// HierarchicalStateMemory_64321430225014
// MI455X (gfx1250) — matrix operand order repaired and checked by reading, not compiled, not run on hardware
//
#include <hip/hip_runtime.h>

typedef __attribute__((ext_vector_type(16))) __bf16 v16bf;
typedef __attribute__((ext_vector_type(8)))  __bf16 v8bf;
typedef __attribute__((ext_vector_type(8)))  float  v8f;
typedef __attribute__((ext_vector_type(4)))  float  v4f;
typedef __attribute__((ext_vector_type(4)))  unsigned uint4v;
typedef __attribute__((ext_vector_type(8)))  int     int8v;
typedef __attribute__((ext_vector_type(4)))  int     int4v;

#define B_  8
#define C_  384
#define T_  4096
#define C3_ 1152
#define NCH 64   /* scan chunks */
#define CH  64   /* chunk length */
#define G_  (B_ * C_)  /* 3072 channels */

__device__ __forceinline__ float sigf(float x) { return 1.f / (1.f + __expf(-x)); }
__device__ __forceinline__ float tanhfast(float x) { return 2.f * sigf(2.f * x) - 1.f; }

// ---- TDM: async 2D tile load (bf16) global -> LDS with row padding -------------------
// tile_x elements per row (bf16), tile_y rows, row stride = stride_elems (bf16).
// LDS rows padded: 64 dwords (128 bf16) data + 4 dwords (8 bf16) pad -> pitch 136 bf16.
__device__ __forceinline__ void tdm_load_tile_bf16(unsigned lds_off, const void* gptr,
                                                   unsigned tile_x, unsigned tile_y,
                                                   unsigned stride_elems) {
  unsigned long long ga = (unsigned long long)(uintptr_t)gptr;
  uint4v g0;
  g0[0] = 1u;                                   // count=1 (valid user descriptor)
  g0[1] = lds_off;                              // lds_addr (bytes)
  g0[2] = (unsigned)ga;                         // global_addr[31:0]
  g0[3] = (unsigned)(ga >> 32) | (2u << 30);    // global_addr[56:32] | type=2 (image)
  int8v g1;
  g1[0] = (int)((1u << 16)      // data_size = 1 -> 2 bytes
              | (1u << 20)      // pad_enable
              | (5u << 22)      // pad_interval: 64 dwords
              | (3u << 25));    // pad_amount: 4 dwords
  g1[1] = (int)(tile_x << 16);                          // tensor_dim0[15:0]
  g1[2] = (int)((tile_x >> 16) | (tile_y << 16));       // tensor_dim0 hi / tensor_dim1 lo
  g1[3] = (int)((tile_y >> 16) | (tile_x << 16));       // tensor_dim1 hi / tile_dim0
  g1[4] = (int)(tile_y);                                // tile_dim1 (tile_dim2 = 0)
  g1[5] = (int)(stride_elems);                          // tensor_dim0_stride[31:0]
  g1[6] = 0;
  g1[7] = 0;
  int4v z4 = {0, 0, 0, 0};
  int8v z8 = {0, 0, 0, 0, 0, 0, 0, 0};
  __builtin_amdgcn_tensor_load_to_lds(g0, g1, z4, z4, z8, 0);
}

// ---------------- Weight transpose: W f32 [N][K] -> Wt bf16 [K][N] ----------------
__global__ __launch_bounds__(256) void hsm_wt_kernel(const float* __restrict__ W,
                                                     __bf16* __restrict__ Wt,
                                                     int N, int K) {
  int e = blockIdx.x * 256 + threadIdx.x;
  if (e >= N * K) return;
  int n = e / K, k = e % K;
  Wt[(size_t)k * N + n] = (__bf16)W[e];
}

// ---------------- x transpose: x f32 [B][C][T] -> Xt bf16 [(b*T+t)][C] ----------------
__global__ __launch_bounds__(256) void hsm_tx_kernel(const float* __restrict__ x,
                                                     __bf16* __restrict__ Xt) {
  __shared__ float tile[32][65];
  int t0 = blockIdx.x * 64, c0 = blockIdx.y * 32, b = blockIdx.z;
  for (int e = threadIdx.x; e < 2048; e += 256) {
    int i = e >> 6, j = e & 63;
    tile[i][j] = x[(size_t)b * (C_ * T_) + (size_t)(c0 + i) * T_ + (t0 + j)];
  }
  __syncthreads();
  for (int e = threadIdx.x; e < 2048; e += 256) {
    int tt = e >> 5, cc = e & 31;
    Xt[((size_t)b * T_ + t0 + tt) * C_ + (c0 + cc)] = (__bf16)tile[cc][tt];
  }
}

// ------- Generic bf16 WMMA GEMM, TDM double-buffered B: out[M][N] = A[M][K]*Wt[K][N] ----
__global__ __launch_bounds__(256) void hsm_gemm_bf16(const __bf16* __restrict__ A,
                                                     const __bf16* __restrict__ Wt,
                                                     __bf16* __restrict__ out,
                                                     int M, int N, int K) {
  __shared__ __bf16 As[16 * 392];
  __shared__ __bf16 Bs[2][32 * 136];
  const int tid = threadIdx.x;
  const int wave = tid >> 5, lane = tid & 31;
  const int m0 = blockIdx.x * 16;
  const int n0 = blockIdx.y * 128;
  const int mlane = lane & 15, half = lane >> 4;
  const unsigned bs_off0 = (unsigned)(uintptr_t)(void*)&Bs[0][0];
  const unsigned bs_off1 = (unsigned)(uintptr_t)(void*)&Bs[1][0];

  // Prologue: issue TDM for tile 0 while A is staged by all waves.
  if (wave == 0)
    tdm_load_tile_bf16(bs_off0, &Wt[n0], 128, 32, (unsigned)N);

  for (int e = tid; e < 16 * 384; e += 256) {
    int r = e / 384, k = e % 384;
    int m = m0 + r;
    As[r * 392 + k] = (m < M) ? A[(size_t)m * K + k] : (__bf16)0.f;
  }

  v8f acc = {};
  const int ksteps = K >> 5;
  for (int kk = 0; kk < ksteps; ++kk) {
    if (wave == 0) {
      if (kk + 1 < ksteps) {
        // issue next tile into the other buffer, then require only the oldest done
        tdm_load_tile_bf16((kk & 1) ? bs_off0 : bs_off1,
                           &Wt[(size_t)((kk + 1) * 32) * N + n0], 128, 32, (unsigned)N);
        __builtin_amdgcn_s_wait_tensorcnt(1);
      } else {
        __builtin_amdgcn_s_wait_tensorcnt(0);
      }
    }
    __syncthreads();  // tile kk visible to all waves (also covers As on first iter)

    const __bf16* bs = &Bs[kk & 1][0];
    v16bf afrag, bfrag;
    *(v8bf*)&afrag         = *(const v8bf*)&As[mlane * 392 + kk * 32 + half * 8];
    *(((v8bf*)&afrag) + 1) = *(const v8bf*)&As[mlane * 392 + kk * 32 + 16 + half * 8];
    for (int s = 0; s < 8; ++s) {
      bfrag[s]     = bs[(half * 8 + s) * 136 + wave * 16 + mlane];
      bfrag[s + 8] = bs[(16 + half * 8 + s) * 136 + wave * 16 + mlane];
    }

    acc = __builtin_amdgcn_wmma_f32_16x16x32_bf16(false, afrag, false, bfrag,
                                                  (short)0, acc, false, false);
    __syncthreads();  // all waves done with buffer (kk&1) before it is re-targeted
  }

  const int n = n0 + wave * 16 + mlane;
#pragma unroll
  for (int r = 0; r < 8; ++r) {
    int m = m0 + half * 8 + r;
    if (m < M) out[(size_t)m * N + n] = (__bf16)acc[r];
  }
}

// ======================= Level-0 recurrence as a chunked parallel scan ==================
// s_t = a_t * s_{t-1} + b_t ;  a = sig(f), b = sig(i)*tanh(c)

__global__ __launch_bounds__(256) void hsm_rec0_phase1(const __bf16* __restrict__ P,
                                                       __bf16* __restrict__ a_all,
                                                       __bf16* __restrict__ b_all,
                                                       float* __restrict__ chunkA,
                                                       float* __restrict__ chunkB) {
  int th = blockIdx.x * 256 + threadIdx.x;       // 0 .. NCH*G_-1
  int ch = th / G_, g = th % G_;
  int b = g / C_, c = g % C_;
  float A = 1.f, Bv = 0.f;
  int t0 = ch * CH;
  for (int i = 0; i < CH; ++i) {
    int t = t0 + i;
    size_t row = ((size_t)b * T_ + t) * C3_;
    float av = sigf((float)P[row + c]);
    float bv = sigf((float)P[row + C_ + c]) * tanhfast((float)P[row + 2 * C_ + c]);
    a_all[(size_t)t * G_ + g] = (__bf16)av;
    b_all[(size_t)t * G_ + g] = (__bf16)bv;
    A *= av;
    Bv = av * Bv + bv;
  }
  chunkA[(size_t)ch * G_ + g] = A;
  chunkB[(size_t)ch * G_ + g] = Bv;
}

__global__ __launch_bounds__(128) void hsm_rec0_phase2(const float* __restrict__ chunkA,
                                                       const float* __restrict__ chunkB,
                                                       const float* __restrict__ s_in,
                                                       float* __restrict__ carry,
                                                       __bf16* __restrict__ s_commit,
                                                       float* __restrict__ s_out) {
  int g = blockIdx.x * 128 + threadIdx.x;
  int b = g / C_, c = g % C_;
  float s = s_in[g];
  for (int ch = 0; ch < NCH; ++ch) {
    carry[(size_t)ch * G_ + g] = s;
    s = chunkA[(size_t)ch * G_ + g] * s + chunkB[(size_t)ch * G_ + g];
    s_commit[(size_t)(ch * 8 + b) * C_ + c] = (__bf16)s;   // s0 at t = 64*ch+63
  }
  s_out[g] = s;
}

__global__ __launch_bounds__(256) void hsm_rec0_phase3(const __bf16* __restrict__ a_all,
                                                       const __bf16* __restrict__ b_all,
                                                       const float* __restrict__ carry,
                                                       __bf16* __restrict__ s_all) {
  int th = blockIdx.x * 256 + threadIdx.x;
  int ch = th / G_, g = th % G_;
  int b = g / C_, c = g % C_;
  float s = carry[(size_t)ch * G_ + g];
  int t0 = ch * CH;
  for (int i = 0; i < CH; ++i) {
    int t = t0 + i;
    s = (float)a_all[(size_t)t * G_ + g] * s + (float)b_all[(size_t)t * G_ + g];
    s_all[((size_t)b * T_ + t) * C_ + c] = (__bf16)s;
  }
}

// ---------------- Level-1 recurrence over 64 commits ----------------
__global__ __launch_bounds__(128) void hsm_rec1(const __bf16* __restrict__ Q,
                                                const float* __restrict__ s_in,
                                                __bf16* __restrict__ s_all,
                                                float* __restrict__ s_out) {
  int g = blockIdx.x * 128 + threadIdx.x;
  int b = g / C_, c = g % C_;
  float s = s_in[g];
  s_all[(size_t)b * C_ + c] = (__bf16)s;  // k=0
  for (int k = 1; k <= 64; ++k) {
    size_t row = ((size_t)(k - 1) * 8 + b) * C3_;
    float f = (float)Q[row + c];
    float i = (float)Q[row + C_ + c];
    float cc = (float)Q[row + 2 * C_ + c];
    s = sigf(f) * s + sigf(i) * tanhfast(cc);
    s_all[((size_t)k * 8 + b) * C_ + c] = (__bf16)s;
  }
  s_out[g] = s;
}

// ---------------- Level-2: single update ----------------
__global__ __launch_bounds__(128) void hsm_rec2(const __bf16* __restrict__ Q,
                                                const float* __restrict__ s_in,
                                                __bf16* __restrict__ s_bf,
                                                float* __restrict__ s_out) {
  int g = blockIdx.x * 128 + threadIdx.x;
  int b = g / C_, c = g % C_;
  size_t row = (size_t)b * C3_;
  float f = (float)Q[row + c];
  float i = (float)Q[row + C_ + c];
  float cc = (float)Q[row + 2 * C_ + c];
  float s = sigf(f) * s_in[g] + sigf(i) * tanhfast(cc);
  s_out[g] = s;
  s_bf[g] = (__bf16)s;
}

// ---------------- Output GEMM with fused window terms, t-contiguous stores ----------------
__global__ __launch_bounds__(256) void hsm_gemm_y(const __bf16* __restrict__ A,
                                                  const __bf16* __restrict__ Wt,
                                                  const __bf16* __restrict__ U1,
                                                  const __bf16* __restrict__ U2,
                                                  float* __restrict__ y) {
  __shared__ __bf16 As[16 * 392];
  __shared__ __bf16 Bs[2][32 * 136];
  __shared__ float Ds[8 * 16 * 17];
  const int tid = threadIdx.x;
  const int wave = tid >> 5, lane = tid & 31;
  const int m0 = blockIdx.x * 16;   // = b*T + t0
  const int n0 = blockIdx.y * 128;
  const int mlane = lane & 15, half = lane >> 4;
  const int K = C_, N = C_;
  const unsigned bs_off0 = (unsigned)(uintptr_t)(void*)&Bs[0][0];
  const unsigned bs_off1 = (unsigned)(uintptr_t)(void*)&Bs[1][0];

  if (wave == 0)
    tdm_load_tile_bf16(bs_off0, &Wt[n0], 128, 32, (unsigned)N);

  for (int e = tid; e < 16 * 384; e += 256) {
    int r = e / 384, k = e % 384;
    As[r * 392 + k] = A[(size_t)(m0 + r) * K + k];
  }

  v8f acc = {};
  const int ksteps = K >> 5;
  for (int kk = 0; kk < ksteps; ++kk) {
    if (wave == 0) {
      if (kk + 1 < ksteps) {
        tdm_load_tile_bf16((kk & 1) ? bs_off0 : bs_off1,
                           &Wt[(size_t)((kk + 1) * 32) * N + n0], 128, 32, (unsigned)N);
        __builtin_amdgcn_s_wait_tensorcnt(1);
      } else {
        __builtin_amdgcn_s_wait_tensorcnt(0);
      }
    }
    __syncthreads();

    const __bf16* bs = &Bs[kk & 1][0];
    v16bf afrag, bfrag;
    *(v8bf*)&afrag         = *(const v8bf*)&As[mlane * 392 + kk * 32 + half * 8];
    *(((v8bf*)&afrag) + 1) = *(const v8bf*)&As[mlane * 392 + kk * 32 + 16 + half * 8];
    for (int s = 0; s < 8; ++s) {
      bfrag[s]     = bs[(half * 8 + s) * 136 + wave * 16 + mlane];
      bfrag[s + 8] = bs[(16 + half * 8 + s) * 136 + wave * 16 + mlane];
    }
    acc = __builtin_amdgcn_wmma_f32_16x16x32_bf16(false, afrag, false, bfrag,
                                                  (short)0, acc, false, false);
    __syncthreads();
  }

  float* dsw = &Ds[wave * 16 * 17];
#pragma unroll
  for (int r = 0; r < 8; ++r)
    dsw[(half * 8 + r) * 17 + mlane] = acc[r];

  const int b = m0 >> 12;
  const int t0 = m0 & (T_ - 1);
  const int nrow = lane & 15;
  const int th = lane >> 4;
  const int n = n0 + wave * 16 + nrow;

  float vals[8];
#pragma unroll
  for (int j = 0; j < 8; ++j) {
    int t = t0 + th * 8 + j;
    float v = dsw[(th * 8 + j) * 17 + nrow];
    int k = (t + 1) >> 6;
    v += (float)U1[(size_t)(k * 8 + b) * C_ + n];
    if (t == T_ - 1) v += (float)U2[(size_t)b * C_ + n];
    vals[j] = v;
  }
  float* dst = y + ((size_t)b * C_ + n) * T_ + t0 + th * 8;
  ((v4f*)dst)[0] = (v4f){vals[0], vals[1], vals[2], vals[3]};
  ((v4f*)dst)[1] = (v4f){vals[4], vals[5], vals[6], vals[7]};
}

extern "C" void kernel_launch(void* const* d_in, const int* in_sizes, int n_in,
                              void* d_out, int out_size, void* d_ws, size_t ws_size,
                              hipStream_t stream) {
  (void)in_sizes; (void)n_in; (void)out_size; (void)ws_size;
  const float* x     = (const float*)d_in[0];
  const float* W0    = (const float*)d_in[1];
  const float* W1    = (const float*)d_in[2];
  const float* W2    = (const float*)d_in[3];
  const float* Wout  = (const float*)d_in[4];
  const float* s0_in = (const float*)d_in[5];
  const float* s1_in = (const float*)d_in[6];
  const float* s2_in = (const float*)d_in[7];

  float* y      = (float*)d_out;
  float* s0_out = y + (size_t)B_ * C_ * T_;
  float* s1_out = s0_out + B_ * C_;
  float* s2_out = s1_out + B_ * C_;

  size_t cur = 0;
  char* wsb = (char*)d_ws;
  auto carve = [&](size_t bytes) -> void* {
    void* p = wsb + cur;
    cur += (bytes + 255) & ~(size_t)255;
    return p;
  };

  __bf16* Xt     = (__bf16*)carve((size_t)B_ * T_ * C_ * 2);
  __bf16* P0     = (__bf16*)carve((size_t)B_ * T_ * C3_ * 2);
  __bf16* s0_all = (__bf16*)carve((size_t)B_ * T_ * C_ * 2);
  __bf16* a_all  = (__bf16*)carve((size_t)T_ * G_ * 2);
  __bf16* b_all  = (__bf16*)carve((size_t)T_ * G_ * 2);
  float*  chunkA = (float*)carve((size_t)NCH * G_ * 4);
  float*  chunkB = (float*)carve((size_t)NCH * G_ * 4);
  float*  carry  = (float*)carve((size_t)NCH * G_ * 4);
  __bf16* s0c    = (__bf16*)carve((size_t)512 * C_ * 2);
  __bf16* Q1     = (__bf16*)carve((size_t)512 * C3_ * 2);
  __bf16* s1_all = (__bf16*)carve((size_t)520 * C_ * 2);
  __bf16* Q2     = (__bf16*)carve((size_t)16 * C3_ * 2);
  __bf16* s2bf   = (__bf16*)carve((size_t)16 * C_ * 2);
  __bf16* U1     = (__bf16*)carve((size_t)528 * C_ * 2);
  __bf16* U2     = (__bf16*)carve((size_t)16 * C_ * 2);
  __bf16* Wt0    = (__bf16*)carve((size_t)C3_ * C_ * 2);
  __bf16* Wt1    = (__bf16*)carve((size_t)C3_ * C_ * 2);
  __bf16* Wt2    = (__bf16*)carve((size_t)C3_ * C_ * 2);
  __bf16* WtO    = (__bf16*)carve((size_t)C3_ * C_ * 2);

  const int wblk = (C3_ * C_ + 255) / 256;
  hsm_wt_kernel<<<wblk, 256, 0, stream>>>(W0, Wt0, C3_, C_);
  hsm_wt_kernel<<<wblk, 256, 0, stream>>>(W1, Wt1, C3_, C_);
  hsm_wt_kernel<<<wblk, 256, 0, stream>>>(W2, Wt2, C3_, C_);
  hsm_wt_kernel<<<wblk, 256, 0, stream>>>(Wout, WtO, C_, C3_);

  hsm_tx_kernel<<<dim3(T_ / 64, C_ / 32, B_), 256, 0, stream>>>(x, Xt);

  // P0 = Xt @ W0^T : [32768, 1152]
  hsm_gemm_bf16<<<dim3(B_ * T_ / 16, C3_ / 128), 256, 0, stream>>>(
      Xt, Wt0, P0, B_ * T_, C3_, C_);

  // level-0 parallel scan
  hsm_rec0_phase1<<<(NCH * G_) / 256, 256, 0, stream>>>(P0, a_all, b_all, chunkA, chunkB);
  hsm_rec0_phase2<<<G_ / 128, 128, 0, stream>>>(chunkA, chunkB, s0_in, carry, s0c, s0_out);
  hsm_rec0_phase3<<<(NCH * G_) / 256, 256, 0, stream>>>(a_all, b_all, carry, s0_all);

  // Q1 = s0_commit @ W1^T : [512, 1152]
  hsm_gemm_bf16<<<dim3(512 / 16, C3_ / 128), 256, 0, stream>>>(
      s0c, Wt1, Q1, 512, C3_, C_);

  hsm_rec1<<<G_ / 128, 128, 0, stream>>>(Q1, s1_in, s1_all, s1_out);

  // Q2 = s1_final @ W2^T : [8, 1152]
  hsm_gemm_bf16<<<dim3(1, C3_ / 128), 256, 0, stream>>>(
      s1_all + (size_t)64 * 8 * C_, Wt2, Q2, 8, C3_, C_);

  hsm_rec2<<<G_ / 128, 128, 0, stream>>>(Q2, s2_in, s2bf, s2_out);

  // U1[k] = s1_k @ WoutB^T : [520, 384]
  hsm_gemm_bf16<<<dim3(33, C_ / 128), 256, 0, stream>>>(
      s1_all, WtO + (size_t)384 * C_, U1, 520, C_, C_);

  // U2 = s2 @ WoutC^T : [8, 384]
  hsm_gemm_bf16<<<dim3(1, C_ / 128), 256, 0, stream>>>(
      s2bf, WtO + (size_t)768 * C_, U2, 8, C_, C_);

  // y = s0_all @ WoutA^T + U1[window] (+ U2 at t = T-1)
  hsm_gemm_y<<<dim3(B_ * T_ / 16, C_ / 128), 256, 0, stream>>>(
      s0_all, WtO, U1, U2, y);
}
